// Prior_41343355191396
// MI455X (gfx1250) — matrix operand order repaired and checked by reading, not compiled, not run on hardware
//
#include <hip/hip_runtime.h>
#include <hip/hip_bf16.h>

// DMM gated-transition prior, 1000-step recurrence.
// 256 batch rows are INDEPENDENT recurrences -> 16 workgroups of 16 rows
// (one WMMA M-tile) each, no inter-WG sync. All weights fp16 in LDS
// (~319KB; CDNA5 allows 320KB/WG). Per step (4 barriers):
//   A: hg=relu(z@Wg1^T+bg1), hp=relu(z@Wp1^T+bp1)          (26 tiles)
//   B: ag=hg@Wg2^T+bg2, pm=hp@Wp2^T+bp2                    (14 tiles)
//   C: as=relu(pm)@Wsig^T+bsig, zloc=z@Wloc^T+bloc         (14 tiles)
//   D: g=sigmoid(ag); loc=(1-g)*zloc+g*pm; scale=softplus(as);
//      z=loc+scale*eps; vectorized float4 NT stores of z/loc/scale
// GEMMs via v_wmma_f32_16x16x32_f16 (fp16 in, fp32 accumulate).

typedef __attribute__((ext_vector_type(16))) _Float16 v16h;
typedef __attribute__((ext_vector_type(8)))  _Float16 v8h;
typedef __attribute__((ext_vector_type(4)))  _Float16 h4;
typedef __attribute__((ext_vector_type(8)))  float    v8f;
typedef __attribute__((ext_vector_type(4)))  float    f4;

#define T_STEPS 1000
#define NGEN    256
#define ZD      100   // z dim
#define HD      200   // hidden dim
#define ZP      128   // z padded as K (multiple of 32)
#define ZT      112   // z padded as N (7 tiles of 16)
#define HP      224   // h padded as K (multiple of 32)
#define HT      208   // h padded as N (13 tiles of 16)
#define NW      16    // waves per block
#define NTHREADS (NW*32)
#define NQ      (ZT/4)   // 28 quads per row in phase D

// One 16x16 output tile of  C = A(16xK) * B^T + bias,  fp16 inputs, f32 acc.
// A: row-major [16][sa] fp16 in LDS. W: weights row-major [out][sb] fp16 in
// LDS (so the WMMA B-matrix, K x 16, is two runs of 8 contiguous K per lane, 16 apart).
__device__ __forceinline__ v8f wmma_tile(const _Float16* __restrict__ A, int sa,
                                         const _Float16* __restrict__ W, int sb,
                                         int ntile, int kiters,
                                         const float* __restrict__ bias, int lane) {
  const int nloc = lane & 15;
  const int half = lane >> 4;
  const int n    = ntile * 16 + nloc;
  const float bv = bias[n];
  v8f c = {bv, bv, bv, bv, bv, bv, bv, bv};
  // A-frag: lane holds row m=nloc, K = {ko..ko+7, 16+ko..16+ko+7}, ko = 8*half
  const _Float16* ap = A + nloc * sa + half * 8;
  // B-frag: lane holds col n, K = {ko..ko+7, 16+ko..16+ko+7}, ko = 8*half
  const _Float16* bp = W + n * sb + half * 8;
#pragma unroll
  for (int kb = 0; kb < kiters; ++kb) {
    v8h alo = *(const v8h*)(ap);
    v8h ahi = *(const v8h*)(ap + 16);
    v8h blo = *(const v8h*)(bp);
    v8h bhi = *(const v8h*)(bp + 16);
    v16h af = __builtin_shufflevector(alo, ahi, 0,1,2,3,4,5,6,7,8,9,10,11,12,13,14,15);
    v16h bf = __builtin_shufflevector(blo, bhi, 0,1,2,3,4,5,6,7,8,9,10,11,12,13,14,15);
    c = __builtin_amdgcn_wmma_f32_16x16x32_f16(false, af, false, bf, (short)0, c, false, false);
    ap += 32; bp += 32;
  }
  return c;
}

__global__ __launch_bounds__(NTHREADS, 1)
void prior_recurrence_kernel(const float* __restrict__ eps,
                             const float* __restrict__ zq0,
                             const float* __restrict__ Wg1, const float* __restrict__ bg1,
                             const float* __restrict__ Wg2, const float* __restrict__ bg2,
                             const float* __restrict__ Wp1, const float* __restrict__ bp1,
                             const float* __restrict__ Wp2, const float* __restrict__ bp2,
                             const float* __restrict__ Wsig, const float* __restrict__ bsig,
                             const float* __restrict__ Wloc, const float* __restrict__ bloc,
                             float* __restrict__ out) {
  // ---- LDS: weights (fp16, zero-padded) + activations ----
  __shared__ alignas(16) _Float16 sWg1[HT * ZP];   // [208][128]
  __shared__ alignas(16) _Float16 sWp1[HT * ZP];
  __shared__ alignas(16) _Float16 sWg2[ZT * HP];   // [112][224]
  __shared__ alignas(16) _Float16 sWp2[ZT * HP];
  __shared__ alignas(16) _Float16 sWsig[ZT * ZP];  // [112][128]
  __shared__ alignas(16) _Float16 sWloc[ZT * ZP];
  __shared__ alignas(16) _Float16 sZ[16 * ZP];       // z state (A for phases A,C)
  __shared__ alignas(16) _Float16 sHg[16 * HP];      // relu(g1) (A for phase B)
  __shared__ alignas(16) _Float16 sHp[16 * HP];      // relu(p1)
  __shared__ alignas(16) _Float16 sPmRelu[16 * ZP];  // relu(pm) (A for phase C)
  __shared__ alignas(16) float sZloc[16 * ZT];
  __shared__ alignas(16) float sPm[16 * ZT];
  __shared__ alignas(16) float sAg[16 * ZT];
  __shared__ alignas(16) float sAs[16 * ZT];
  __shared__ alignas(16) float sBg1[HT], sBp1[HT];
  __shared__ alignas(16) float sBg2[ZT], sBp2[ZT], sBsig[ZT], sBloc[ZT];

  const int tid  = threadIdx.x;
  const int lane = tid & 31;
  const int w    = tid >> 5;
  const int row0 = blockIdx.x * 16;

  // ---- one-time: load weights f32 -> fp16 with zero padding ----
  for (int i = tid; i < HT * ZP; i += NTHREADS) {
    int r = i >> 7, c = i & (ZP - 1);
    bool ok = (r < HD) && (c < ZD);
    sWg1[i] = (_Float16)(ok ? Wg1[r * ZD + c] : 0.f);
    sWp1[i] = (_Float16)(ok ? Wp1[r * ZD + c] : 0.f);
  }
  for (int i = tid; i < ZT * HP; i += NTHREADS) {
    int r = i / HP, c = i - r * HP;
    bool ok = (r < ZD) && (c < HD);
    sWg2[i] = (_Float16)(ok ? Wg2[r * HD + c] : 0.f);
    sWp2[i] = (_Float16)(ok ? Wp2[r * HD + c] : 0.f);
  }
  for (int i = tid; i < ZT * ZP; i += NTHREADS) {
    int r = i >> 7, c = i & (ZP - 1);
    bool ok = (r < ZD) && (c < ZD);
    sWsig[i] = (_Float16)(ok ? Wsig[r * ZD + c] : 0.f);
    sWloc[i] = (_Float16)(ok ? Wloc[r * ZD + c] : 0.f);
  }
  for (int i = tid; i < HT; i += NTHREADS) {
    sBg1[i] = (i < HD) ? bg1[i] : 0.f;
    sBp1[i] = (i < HD) ? bp1[i] : 0.f;
  }
  for (int i = tid; i < ZT; i += NTHREADS) {
    sBg2[i]  = (i < ZD) ? bg2[i]  : 0.f;
    sBp2[i]  = (i < ZD) ? bp2[i]  : 0.f;
    sBsig[i] = (i < ZD) ? bsig[i] : 0.f;
    sBloc[i] = (i < ZD) ? bloc[i] : 0.f;
  }
  for (int i = tid; i < 16 * ZP; i += NTHREADS) {   // z0 broadcast + pad zero
    int c = i & (ZP - 1);
    sZ[i]      = (_Float16)((c < ZD) ? zq0[c] : 0.f);
    sPmRelu[i] = (_Float16)0.f;
  }
  for (int i = tid; i < 16 * HP; i += NTHREADS) {
    sHg[i] = (_Float16)0.f;
    sHp[i] = (_Float16)0.f;
  }
  __syncthreads();

  float* __restrict__ pZ   = out;
  float* __restrict__ pLoc = out + (size_t)NGEN * T_STEPS * ZD;
  float* __restrict__ pScl = out + (size_t)2 * NGEN * T_STEPS * ZD;

  const int nloc  = lane & 15;
  const int mbase = (lane >> 4) * 8;

  // phase-D mapping: 448 threads (14 waves) handle one 4-col quad each
  const bool dAct = (tid < 16 * NQ);                 // 448 = 14 full waves
  const int  dm   = tid / NQ;                        // row 0..15
  const int  dq   = tid - dm * NQ;                   // quad 0..27
  const bool dReal = dAct && (dq < ZD / 4);          // quads 0..24 are real cols

  for (int t = 0; t < T_STEPS; ++t) {
    // ---- prefetch this step's eps quad into registers (used in phase D) ----
    f4 er = {0.f, 0.f, 0.f, 0.f};
    if (dReal) {
      const f4* ep = (const f4*)(eps + ((size_t)t * NGEN + row0 + dm) * ZD + dq * 4);
      er = __builtin_nontemporal_load(ep);
    }

    // ---- phase A: hg, hp (relu into fp16) ---- 26 tiles over 16 waves
    for (int tile = w; tile < 26; tile += NW) {
      if (tile < 13) {
        v8f c = wmma_tile(sZ, ZP, sWg1, ZP, tile, ZP / 32, sBg1, lane);
        int n = tile * 16 + nloc;
#pragma unroll
        for (int v = 0; v < 8; ++v)
          sHg[(mbase + v) * HP + n] = (_Float16)fmaxf(c[v], 0.f);
      } else {
        int tt = tile - 13;
        v8f c = wmma_tile(sZ, ZP, sWp1, ZP, tt, ZP / 32, sBp1, lane);
        int n = tt * 16 + nloc;
#pragma unroll
        for (int v = 0; v < 8; ++v)
          sHp[(mbase + v) * HP + n] = (_Float16)fmaxf(c[v], 0.f);
      }
    }
    __syncthreads();

    // ---- phase B: ag = hg@Wg2^T, pm = hp@Wp2^T ---- 14 tiles
    if (w < 14) {
      if (w < 7) {
        v8f c = wmma_tile(sHg, HP, sWg2, HP, w, HP / 32, sBg2, lane);
        int n = w * 16 + nloc;
#pragma unroll
        for (int v = 0; v < 8; ++v)
          sAg[(mbase + v) * ZT + n] = c[v];
      } else {
        int tt = w - 7;
        v8f c = wmma_tile(sHp, HP, sWp2, HP, tt, HP / 32, sBp2, lane);
        int n = tt * 16 + nloc;
#pragma unroll
        for (int v = 0; v < 8; ++v) {
          float pv = c[v];
          sPm[(mbase + v) * ZT + n]     = pv;
          sPmRelu[(mbase + v) * ZP + n] = (_Float16)fmaxf(pv, 0.f);
        }
      }
    }
    __syncthreads();

    // ---- phase C: as = relu(pm)@Wsig^T, zloc = z@Wloc^T ---- 14 tiles
    if (w < 14) {
      if (w < 7) {
        v8f c = wmma_tile(sPmRelu, ZP, sWsig, ZP, w, ZP / 32, sBsig, lane);
        int n = w * 16 + nloc;
#pragma unroll
        for (int v = 0; v < 8; ++v)
          sAs[(mbase + v) * ZT + n] = c[v];
      } else {
        int tt = w - 7;
        v8f c = wmma_tile(sZ, ZP, sWloc, ZP, tt, ZP / 32, sBloc, lane);
        int n = tt * 16 + nloc;
#pragma unroll
        for (int v = 0; v < 8; ++v)
          sZloc[(mbase + v) * ZT + n] = c[v];
      }
    }
    __syncthreads();

    // ---- phase D: gate, sample, vectorized state + output writes ----
    if (dAct) {
      const int base = dm * ZT + dq * 4;
      f4 zl = *(const f4*)(sZloc + base);
      f4 pm = *(const f4*)(sPm + base);
      f4 ag = *(const f4*)(sAg + base);
      f4 as = *(const f4*)(sAs + base);
      f4 loc, scl, zt;
      h4 zh;
#pragma unroll
      for (int e = 0; e < 4; ++e) {
        float g = 1.f / (1.f + __expf(-ag[e]));
        loc[e]  = (1.f - g) * zl[e] + g * pm[e];
        scl[e]  = (as[e] > 15.f) ? as[e] : log1pf(__expf(as[e]));
        zt[e]   = loc[e] + scl[e] * er[e];
        zh[e]   = (_Float16)zt[e];
      }
      *(h4*)(sZ + dm * ZP + dq * 4) = zh;            // state update (b64)
      if (dReal) {
        size_t o = ((size_t)(row0 + dm) * T_STEPS + t) * ZD + dq * 4;
        __builtin_nontemporal_store(zt,  (f4*)(pZ   + o));
        __builtin_nontemporal_store(loc, (f4*)(pLoc + o));
        __builtin_nontemporal_store(scl, (f4*)(pScl + o));
      }
    }
    __syncthreads();
  }
}

extern "C" void kernel_launch(void* const* d_in, const int* in_sizes, int n_in,
                              void* d_out, int out_size, void* d_ws, size_t ws_size,
                              hipStream_t stream) {
  // setup_inputs order:
  // 0 length, 1 N_generate, 2 eps, 3 z_q_0, 4 Wg1, 5 bg1, 6 Wg2, 7 bg2,
  // 8 Wp1, 9 bp1, 10 Wp2, 11 bp2, 12 Wsig, 13 bsig, 14 Wloc, 15 bloc
  const float* eps  = (const float*)d_in[2];
  const float* zq0  = (const float*)d_in[3];
  const float* Wg1  = (const float*)d_in[4];
  const float* bg1  = (const float*)d_in[5];
  const float* Wg2  = (const float*)d_in[6];
  const float* bg2  = (const float*)d_in[7];
  const float* Wp1  = (const float*)d_in[8];
  const float* bp1  = (const float*)d_in[9];
  const float* Wp2  = (const float*)d_in[10];
  const float* bp2  = (const float*)d_in[11];
  const float* Wsig = (const float*)d_in[12];
  const float* bsig = (const float*)d_in[13];
  const float* Wloc = (const float*)d_in[14];
  const float* bloc = (const float*)d_in[15];
  float* out = (float*)d_out;

  dim3 grid(NGEN / 16);      // 16 independent 16-row recurrences
  dim3 block(NTHREADS);      // 16 waves (wave32)
  prior_recurrence_kernel<<<grid, block, 0, stream>>>(
      eps, zq0, Wg1, bg1, Wg2, bg2, Wp1, bp1, Wp2, bp2, Wsig, bsig, Wloc, bloc, out);
}
